// OnsetEdgePoolingVersion2_82308753260716
// MI455X (gfx1250) — matrix operand order repaired and checked by reading, not compiled, not run on hardware
//
#include <hip/hip_runtime.h>
#include <hip/hip_bf16.h>

// ---------------------------------------------------------------------------
// Problem constants (match reference)
// ---------------------------------------------------------------------------
#define NN 50000      // nodes
#define EE 100000     // edges
#define CC 256        // channels

// WMMA fragment types (CDNA5 wave32)
typedef __attribute__((ext_vector_type(16))) __bf16 v16bf;
typedef __attribute__((ext_vector_type(8)))  float  v8f;

#define WPAD 264      // bf16 elements per W row in LDS (256 + 8 pad vs bank conflicts)
#define ROWT 16       // 16x16 row tiles per block in GEMM

// ---------------------------------------------------------------------------
// Kernel 1: init cnt = 1.0 (self-loop count)
// ---------------------------------------------------------------------------
__global__ void init_cnt_kernel(float* __restrict__ cnt) {
    int i = blockIdx.x * blockDim.x + threadIdx.x;
    if (i < NN) cnt[i] = 1.0f;
}

// ---------------------------------------------------------------------------
// Kernel 2: xt = x @ W^T + b via v_wmma_f32_16x16x32_bf16.
// W (256x256 f32) is converted to bf16 and cached in LDS once per block.
// Each block: 256 threads = 8 waves. Each wave owns two 16-wide column tiles
// (n0 and n0+128) of one 16-row tile; block loops over ROWT row tiles.
// Epilogue writes xt AND sums (= self-loop init of the scatter accumulator).
// ---------------------------------------------------------------------------
__global__ void gemm_wmma_kernel(const float* __restrict__ x,
                                 const float* __restrict__ W,
                                 const float* __restrict__ bias,
                                 float* __restrict__ xt,
                                 float* __restrict__ sums) {
    extern __shared__ __align__(16) char smem[];
    __bf16* wl = (__bf16*)smem;   // [256][WPAD] bf16, row n = output channel

    // Cooperative fill: W[n][k] -> wl[n*WPAD + k] as bf16 (coalesced global reads)
    for (int idx = threadIdx.x; idx < CC * CC; idx += blockDim.x) {
        int n = idx >> 8;
        int k = idx & 255;
        wl[n * WPAD + k] = (__bf16)W[idx];
    }
    __syncthreads();

    const int wave = threadIdx.x >> 5;
    const int lane = threadIdx.x & 31;
    const int lrow = lane & 15;     // row within A tile / column within B,C tiles
    const int hi   = lane >> 4;     // lane half selects K sub-range / M sub-range

    const int n0a = wave * 16;      // column tiles 0..127
    const int n0b = n0a + 128;      // column tiles 128..255
    const int n_a = n0a + lrow;
    const int n_b = n0b + lrow;
    const float biasA = bias[n_a];
    const float biasB = bias[n_b];

    for (int rt = 0; rt < ROWT; ++rt) {
        const int tile = blockIdx.x * ROWT + rt;
        const int m0 = tile * 16;
        if (m0 >= NN) break;

        const int m = m0 + lrow;                       // A row for this lane
        const float* xrow = x + (size_t)m * CC;

        v8f acc0 = {};
        v8f acc1 = {};

        #pragma unroll
        for (int k0 = 0; k0 < CC; k0 += 32) {
            // A fragment, 16x32 bf16: laneLo holds K=k0+0..15,
            // laneHi holds K=k0+16..31 (pairs per VGPR).
            v16bf a;
            const float* xp = xrow + k0 + hi * 16;
            #pragma unroll
            for (int i = 0; i < 8; ++i) {
                a[i]     = (__bf16)xp[i];
                a[8 + i] = (__bf16)xp[8 + i];
            }
            // B fragments from LDS: column n = n0+lrow, laneLo K=k0..k0+15,
            // laneHi K=k0+16..k0+31. 32 contiguous bytes per lane -> ds_load_b128.
            const int kb = k0 + hi * 16;
            const __bf16* bpA = wl + n_a * WPAD + kb;
            const __bf16* bpB = wl + n_b * WPAD + kb;
            v16bf bA, bB;
            #pragma unroll
            for (int i = 0; i < 16; ++i) { bA[i] = bpA[i]; bB[i] = bpB[i]; }

            acc0 = __builtin_amdgcn_wmma_f32_16x16x32_bf16(
                false, a, false, bA, (short)0, acc0, false, false);
            acc1 = __builtin_amdgcn_wmma_f32_16x16x32_bf16(
                false, a, false, bB, (short)0, acc1, false, false);
        }

        // Epilogue: D layout -> VGPR r: laneLo M=m0+r, laneHi M=m0+8+r; N=n0+lrow
        #pragma unroll
        for (int r = 0; r < 8; ++r) {
            const int mm = m0 + r + hi * 8;
            const float vA = acc0[r] + biasA;
            const float vB = acc1[r] + biasB;
            const size_t oA = (size_t)mm * CC + n_a;
            const size_t oB = (size_t)mm * CC + n_b;
            xt[oA]   = vA;
            sums[oA] = vA;   // self-loop contribution pre-seeded
            xt[oB]   = vB;
            sums[oB] = vB;
        }
    }
}

// ---------------------------------------------------------------------------
// Kernel 3: edge scatter. One wave per edge; lane handles 8 coalesced channels.
//   sums[dst] += xt[src];  cnt[dst] += 1
// ---------------------------------------------------------------------------
__global__ void scatter_kernel(const int* __restrict__ ei,
                               const float* __restrict__ xt,
                               float* __restrict__ sums,
                               float* __restrict__ cnt) {
    const int e = blockIdx.x * (blockDim.x >> 5) + (threadIdx.x >> 5);
    if (e >= EE) return;
    const int lane = threadIdx.x & 31;
    const int s = ei[e];
    const int t = ei[EE + e];
    const float* srcp = xt + (size_t)s * CC;
    float* dstp = sums + (size_t)t * CC;
    #pragma unroll
    for (int i = 0; i < 8; ++i) {
        const int c = lane + i * 32;
        atomicAdd(dstp + c, srcp[c]);
    }
    if (lane == 0) atomicAdd(cnt + t, 1.0f);
}

// ---------------------------------------------------------------------------
// Kernel 4: sequential greedy edge merge. Single block. Flag arrays live in
// LDS (2 x 50000 bytes); edge pairs staged into LDS in 1024-edge chunks so the
// serial dependence chain (thread 0) never leaves LDS.
// ---------------------------------------------------------------------------
#define MCHUNK 1024
__global__ void merge_kernel(const int* __restrict__ ei,
                             int* __restrict__ remg) {
    extern __shared__ __align__(16) char smem[];
    unsigned char* rem  = (unsigned char*)smem;           // NN
    unsigned char* disc = rem + 50048;                    // NN
    int* sbuf = (int*)(disc + 50048);                     // MCHUNK
    int* tbuf = sbuf + MCHUNK;                            // MCHUNK

    for (int i = threadIdx.x; i < NN; i += blockDim.x) {
        rem[i]  = 1;
        disc[i] = 0;
    }
    __syncthreads();
    // discarded0 = zeros.at[edge_index[0]].set(1)  (idempotent parallel stores)
    for (int e = threadIdx.x; e < EE; e += blockDim.x) {
        disc[ei[e]] = 1;
    }
    __syncthreads();

    for (int base = 0; base < EE; base += MCHUNK) {
        const int n = min(MCHUNK, EE - base);
        for (int j = threadIdx.x; j < n; j += blockDim.x) {
            sbuf[j] = ei[base + j];
            tbuf[j] = ei[EE + base + j];
        }
        __syncthreads();
        if (threadIdx.x == 0) {
            for (int j = 0; j < n; ++j) {
                const int s = sbuf[j];
                const int t = tbuf[j];
                if (s != t && rem[s] && rem[t] && disc[t]) {
                    rem[s]  = 0;
                    disc[t] = 0;
                }
            }
        }
        __syncthreads();
    }

    for (int i = threadIdx.x; i < NN; i += blockDim.x) {
        remg[i] = (int)rem[i];
    }
}

// ---------------------------------------------------------------------------
// Kernel 5: out = (sums / max(cnt,1)) * mask ; append mask (as float) to d_out
// ---------------------------------------------------------------------------
__global__ void finalize_kernel(const float* __restrict__ sums,
                                const float* __restrict__ cnt,
                                const int* __restrict__ remg,
                                float* __restrict__ out) {
    const size_t idx = (size_t)blockIdx.x * blockDim.x + threadIdx.x;
    if (idx < (size_t)NN * CC) {
        const int i = (int)(idx >> 8);
        const float inv = 1.0f / fmaxf(cnt[i], 1.0f);
        const float msk = remg[i] ? 1.0f : 0.0f;
        out[idx] = sums[idx] * inv * msk;
    }
    if (idx < NN) {
        out[(size_t)NN * CC + idx] = (float)remg[idx];
    }
}

// ---------------------------------------------------------------------------
// Launcher
// ---------------------------------------------------------------------------
extern "C" void kernel_launch(void* const* d_in, const int* in_sizes, int n_in,
                              void* d_out, int out_size, void* d_ws, size_t ws_size,
                              hipStream_t stream) {
    const float* x    = (const float*)d_in[0];  // [NN, CC]
    const int*   ei   = (const int*)  d_in[1];  // [2, EE] flat: src then dst
    const float* W    = (const float*)d_in[2];  // [CC, CC]
    const float* bias = (const float*)d_in[3];  // [CC]
    float* out = (float*)d_out;                 // [NN*CC] then [NN] mask

    // Workspace carve-up
    char* ws = (char*)d_ws;
    float* xt   = (float*)ws;                                   // NN*CC f32
    float* sums = (float*)(ws + (size_t)NN * CC * 4);           // NN*CC f32
    float* cnt  = (float*)(ws + (size_t)NN * CC * 8);           // NN f32
    int*   remg = (int*)  (ws + (size_t)NN * CC * 8 + NN * 4);  // NN i32

    // 1) cnt = 1 (self-loops)
    init_cnt_kernel<<<(NN + 255) / 256, 256, 0, stream>>>(cnt);

    // 2) GEMM with WMMA; writes xt and sums(=xt)
    {
        const int rowTiles = (NN + 15) / 16;                 // 3125
        const int grid = (rowTiles + ROWT - 1) / ROWT;       // 196
        const size_t lds = (size_t)CC * WPAD * sizeof(__bf16); // 135168 B
        gemm_wmma_kernel<<<grid, 256, lds, stream>>>(x, W, bias, xt, sums);
    }

    // 3) edge scatter-add (atomics)
    {
        const int edgesPerBlock = 256 / 32;
        const int grid = (EE + edgesPerBlock - 1) / edgesPerBlock;
        scatter_kernel<<<grid, 256, 0, stream>>>(ei, xt, sums, cnt);
    }

    // 4) sequential greedy merge in LDS
    {
        const size_t lds = 50048ull * 2 + (size_t)MCHUNK * 2 * sizeof(int);
        merge_kernel<<<1, 256, lds, stream>>>(ei, remg);
    }

    // 5) normalize + mask + mask tail
    {
        const size_t total = (size_t)NN * CC;
        const int grid = (int)((total + 255) / 256);
        finalize_kernel<<<grid, 256, 0, stream>>>(sums, cnt, remg, out);
    }
}
